// WindowAttention_68762426408976
// MI455X (gfx1250) — matrix operand order repaired and checked by reading, not compiled, not run on hardware
//
#include <hip/hip_runtime.h>
#include <hip/hip_bf16.h>
#include <stdint.h>

// ---------------------------------------------------------------------------
// MI455X (gfx1250) window-attention:
//   pack(roll+window, f32->bf16) -> QKV GEMM (bf16 WMMA) -> per-window attn
//   (bf16 WMMA scores + wave32 shuffle softmax) -> out GEMM (bf16 WMMA, f32
//   out) -> scatter(un-window + reverse roll).
// Shapes fixed by the reference: B=8, C=8, R=64, P=4, E=512, H=8, dh=64,
// N=4096 windows, M=B*N=32768 rows.
// ---------------------------------------------------------------------------

typedef __attribute__((ext_vector_type(16))) __bf16       bf16x16;
typedef __attribute__((ext_vector_type(8)))  float        v8f;
typedef __attribute__((ext_vector_type(4)))  unsigned int u32x4;
typedef __attribute__((ext_vector_type(2)))  unsigned int u32x2;
typedef __attribute__((ext_vector_type(4)))  float        f32x4;

union FragU { bf16x16 v; u32x4 u[2]; };

// --------------------------- f32 -> bf16 convert ---------------------------
__global__ void cvt_f32_bf16(const float* __restrict__ s,
                             __bf16* __restrict__ d, int n) {
  int i = blockIdx.x * 256 + threadIdx.x;
  if (i < n) d[i] = (__bf16)s[i];
}

// ------------- pack: roll(+4) + window partition + bf16 convert ------------
// xw[m=n*8+b][e=((c*4+px)*4+py)*4+pz] = x[b][c][(4ix+px-4)&63][(4iy+py-4)&63][(4iz+pz-4)&63]
// one thread handles the 4-contiguous pz run (aligned float4 read).
__global__ void pack_kernel(const float* __restrict__ x,
                            __bf16* __restrict__ xw) {
  int t = blockIdx.x * 256 + threadIdx.x;         // 0 .. 32768*128-1
  int ecell = t & 127;                            // e/4
  int m     = t >> 7;
  int c  = ecell >> 4, px = (ecell >> 2) & 3, py = ecell & 3;
  int b  = m & 7,      n  = m >> 3;
  int iz = n & 15,     iy = (n >> 4) & 15,    ix = n >> 8;
  int X  = (4 * ix + px + 60) & 63;
  int Y  = (4 * iy + py + 60) & 63;
  int Zb = (4 * iz + 60) & 63;                    // multiple of 4, no wrap in run
  size_t src = ((((size_t)(b * 8 + c) * 64 + X) * 64 + Y) * 64) + Zb;
  f32x4 v = *(const f32x4*)(x + src);
  union { __bf16 h[4]; u32x2 d2; } o;
  o.h[0] = (__bf16)v.x; o.h[1] = (__bf16)v.y;
  o.h[2] = (__bf16)v.z; o.h[3] = (__bf16)v.w;
  *(u32x2*)(xw + ((size_t)m * 512 + ecell * 4)) = o.d2;
}

// ------------------------------- GEMM (bf16) -------------------------------
// C[m,f] = sum_e A[m,e] * W[f,e] + bias[f]
// A: M x K bf16 row-major (lda=K).  W: F x K bf16 row-major (ldw=K).
// block = 256 threads = 8 waves; block tile 64(M) x 256(N);
// wave tile 32x64 = 2x4 WMMA accumulators of v_wmma_f32_16x16x32_bf16.
template <bool OUT_BF16>
__global__ void gemm_kernel(const __bf16* __restrict__ A,
                            const __bf16* __restrict__ W,
                            const float*  __restrict__ bias,
                            void* __restrict__ Cp, int K, int F) {
  const int lane = threadIdx.x & 31;
  const int wave = threadIdx.x >> 5;
  const int wm   = wave & 1;                      // 2 waves in M
  const int wn   = wave >> 1;                     // 4 waves in N
  const int mBase = blockIdx.x * 64 + wm * 32;
  const int nBase = blockIdx.y * 256 + wn * 64;
  const int l15 = lane & 15, lh = lane >> 4;

  v8f zero = {};
  v8f acc[2][4];
#pragma unroll
  for (int i = 0; i < 2; i++)
#pragma unroll
    for (int j = 0; j < 4; j++) acc[i][j] = zero;

  // A-frag lane base: row = mBase+l15 (+16 per mi), K offset lh*8 (+{0,16})
  const __bf16* aRow = A + (size_t)(mBase + l15) * K + lh * 8;
  // B-frag lane base: row = nBase+l15 (+16 per ni), K offset lh*8 (+{0,16})
  const __bf16* wRow = W + (size_t)(nBase + l15) * K + lh * 8;

  for (int k0 = 0; k0 < K; k0 += 32) {
    FragU a[2], b[4];
#pragma unroll
    for (int i = 0; i < 2; i++) {
      const __bf16* p = aRow + (size_t)i * 16 * K + k0;
      a[i].u[0] = *(const u32x4*)p;
      a[i].u[1] = *(const u32x4*)(p + 16);
    }
#pragma unroll
    for (int j = 0; j < 4; j++) {
      const __bf16* p = wRow + (size_t)j * 16 * K + k0;
      b[j].u[0] = *(const u32x4*)p;
      b[j].u[1] = *(const u32x4*)(p + 16);
    }
    if (k0 + 32 < K) {                            // hint next K chunk
      __builtin_prefetch(aRow + k0 + 32, 0, 0);   // global_prefetch_b8
      __builtin_prefetch(wRow + k0 + 32, 0, 0);
    }
#pragma unroll
    for (int i = 0; i < 2; i++)
#pragma unroll
      for (int j = 0; j < 4; j++)
        acc[i][j] = __builtin_amdgcn_wmma_f32_16x16x32_bf16(
            false, a[i].v, false, b[j].v, (short)0, acc[i][j], false, false);
  }

  // epilogue: D[g] lane L -> C[mTile + g + 8*(L>=16)][nTile + (L&15)]
#pragma unroll
  for (int i = 0; i < 2; i++) {
#pragma unroll
    for (int j = 0; j < 4; j++) {
      int col = nBase + j * 16 + l15;
      float bv = bias[col];
#pragma unroll
      for (int g = 0; g < 8; g++) {
        int row = mBase + i * 16 + g + lh * 8;
        float val = acc[i][j][g] + bv;
        if (OUT_BF16)
          ((__bf16*)Cp)[(size_t)row * F + col] = (__bf16)val;
        else
          ((float*)Cp)[(size_t)row * F + col] = val;
      }
    }
  }
}

// ------------------------------- attention ---------------------------------
// One wave per (window n, head h).  qkv: (32768 x 1536) bf16, row m = n*8+b;
// Q cols [0,512), K cols [512,1024), V cols [1024,1536); head slice h*64.
// S = (Q K^T) * 1/sqrt(64) via 2x v_wmma_f32_16x16x32_bf16 (rows 8..15 zero),
// softmax over 8 lanes, O = P V via shuffles + FMA.  O: (32768 x 512) bf16.
__global__ void attn_kernel(const __bf16* __restrict__ qkv,
                            __bf16* __restrict__ O) {
  const int lane = threadIdx.x & 31;
  const int wave = threadIdx.x >> 5;
  const int unit = blockIdx.x * 8 + wave;   // n*8 + h
  const int n = unit >> 3, h = unit & 7;
  const int m0 = n * 8;
  const int c0 = h * 64;
  const int l15 = lane & 15, lh = lane >> 4;

  const u32x4 z4 = {0u, 0u, 0u, 0u};
  v8f acc = {};
#pragma unroll
  for (int kk = 0; kk < 2; kk++) {          // d chunks 0..31, 32..63
    const int d0 = kk * 32;
    FragU qa, kb;
    if (l15 < 8) {
      const __bf16* pq = qkv + (size_t)(m0 + l15) * 1536 + c0 + d0 + lh * 8;
      qa.u[0] = *(const u32x4*)pq;
      qa.u[1] = *(const u32x4*)(pq + 16);
      const __bf16* pk = qkv + (size_t)(m0 + l15) * 1536 + 512 + c0 + d0 + lh * 8;
      kb.u[0] = *(const u32x4*)pk;
      kb.u[1] = *(const u32x4*)(pk + 16);
    } else {
      qa.u[0] = z4; qa.u[1] = z4; kb.u[0] = z4; kb.u[1] = z4;
    }
    acc = __builtin_amdgcn_wmma_f32_16x16x32_bf16(
        false, qa.v, false, kb.v, (short)0, acc, false, false);
  }

  // acc[g] at lane L = S[g + 8*(L>=16)][L&15]; valid region: lanes 0..7, g=i.
  float p[8];
#pragma unroll
  for (int g = 0; g < 8; g++) {
    float s = acc[g] * 0.125f;                       // 1/sqrt(dh)
    float mx = s;
    mx = fmaxf(mx, __shfl_xor(mx, 1, 32));
    mx = fmaxf(mx, __shfl_xor(mx, 2, 32));
    mx = fmaxf(mx, __shfl_xor(mx, 4, 32));
    float e = __expf(s - mx);
    float sum = e;
    sum += __shfl_xor(sum, 1, 32);
    sum += __shfl_xor(sum, 2, 32);
    sum += __shfl_xor(sum, 4, 32);
    p[g] = (lane < 8) ? (e / sum) : 0.0f;            // mask j>=8 & rows>=8
  }

  // O[i][d] = sum_j P[i][j] * V[j][d];  lane handles d=lane and d=lane+32.
  const __bf16* pv = qkv + (size_t)m0 * 1536 + 1024 + c0;
  float va[8], vb[8];
#pragma unroll
  for (int j = 0; j < 8; j++) {
    va[j] = (float)pv[(size_t)j * 1536 + lane];
    vb[j] = (float)pv[(size_t)j * 1536 + lane + 32];
  }
  float oa[8], ob[8];
#pragma unroll
  for (int i = 0; i < 8; i++) { oa[i] = 0.f; ob[i] = 0.f; }
#pragma unroll
  for (int i = 0; i < 8; i++) {
#pragma unroll
    for (int j = 0; j < 8; j++) {
      float pij = __shfl(p[i], j, 32);               // P[i][j] from lane j
      oa[i] = fmaf(pij, va[j], oa[i]);
      ob[i] = fmaf(pij, vb[j], ob[i]);
    }
  }
#pragma unroll
  for (int i = 0; i < 8; i++) {
    O[(size_t)(m0 + i) * 512 + c0 + lane]      = (__bf16)oa[i];
    O[(size_t)(m0 + i) * 512 + c0 + lane + 32] = (__bf16)ob[i];
  }
}

// --------------- scatter: un-window + reverse roll(-4) to d_out ------------
// out[b,c,x,y,z] = res[m=n*8+b][f=((c*4+px)*4+py)*4+pz]
// with X=(x+4)&63, px=X>>4, ix=X&15 (px-major reassembly), likewise Y,Z.
__global__ void scatter_kernel(const float* __restrict__ res,
                               float* __restrict__ out) {
  size_t t = (size_t)blockIdx.x * 256 + threadIdx.x;   // 0 .. 8*8*64^3-1
  int z = (int)(t & 63), y = (int)((t >> 6) & 63), xx = (int)((t >> 12) & 63);
  int c = (int)((t >> 18) & 7), b = (int)(t >> 21);
  int X = (xx + 4) & 63, Y = (y + 4) & 63, Z = (z + 4) & 63;
  int px = X >> 4, ix = X & 15;
  int py = Y >> 4, iy = Y & 15;
  int pz = Z >> 4, iz = Z & 15;
  int f = ((c * 4 + px) * 4 + py) * 4 + pz;
  int n = (ix * 16 + iy) * 16 + iz;
  int m = n * 8 + b;
  out[t] = res[(size_t)m * 512 + f];
}

// ------------------------------- launcher ----------------------------------
extern "C" void kernel_launch(void* const* d_in, const int* in_sizes, int n_in,
                              void* d_out, int out_size, void* d_ws, size_t ws_size,
                              hipStream_t stream) {
  const float* x      = (const float*)d_in[0];   // (8,8,64,64,64)
  const float* wInF   = (const float*)d_in[1];   // (1536,512)
  const float* bIn    = (const float*)d_in[2];   // (1536,)
  const float* wOutF  = (const float*)d_in[3];   // (512,512)
  const float* bOut   = (const float*)d_in[4];   // (512,)

  char* ws = (char*)d_ws;
  constexpr size_t QKV_OFF  = 0;                               // 96 MiB bf16
  constexpr size_t XW_OFF   = 32768ull * 1536 * 2;             // +32 MiB bf16
  constexpr size_t WIN_OFF  = XW_OFF + 32768ull * 512 * 2;     // +1.5 MiB
  constexpr size_t WOUT_OFF = WIN_OFF + 1536ull * 512 * 2;     // +0.5 MiB

  __bf16* qkv   = (__bf16*)(ws + QKV_OFF);
  __bf16* xw    = (__bf16*)(ws + XW_OFF);
  __bf16* wInB  = (__bf16*)(ws + WIN_OFF);
  __bf16* wOutB = (__bf16*)(ws + WOUT_OFF);
  __bf16* Obuf  = xw;                 // xw dead after GEMM1 -> reuse for O
  float*  res   = (float*)(ws + QKV_OFF); // qkv dead after attn -> f32 result

  // weights to bf16
  cvt_f32_bf16<<<3072, 256, 0, stream>>>(wInF,  wInB,  1536 * 512);
  cvt_f32_bf16<<<1024, 256, 0, stream>>>(wOutF, wOutB, 512 * 512);

  // roll + window partition + bf16
  pack_kernel<<<16384, 256, 0, stream>>>(x, xw);

  // QKV projection: (32768x512) x (512x1536) -> bf16
  gemm_kernel<true><<<dim3(512, 6), 256, 0, stream>>>(xw, wInB, bIn, qkv, 512, 1536);

  // 32768 (window,head) attention units, 8 waves per block
  attn_kernel<<<4096, 256, 0, stream>>>(qkv, Obuf);

  // output projection: (32768x512) x (512x512) -> f32 (+bias)
  gemm_kernel<false><<<dim3(512, 2), 256, 0, stream>>>(Obuf, wOutB, bOut, res, 512, 512);

  // un-window + reverse roll
  scatter_kernel<<<65536, 256, 0, stream>>>(res, (float*)d_out);
}
